// eight_related_context_mapping_81423989997958
// MI455X (gfx1250) — matrix operand order repaired and checked by reading, not compiled, not run on hardware
//
#include <hip/hip_runtime.h>
#include <hip/hip_bf16.h>
#include <math.h>

typedef __attribute__((ext_vector_type(16))) _Float16 v16h;
typedef __attribute__((ext_vector_type(8)))  float    v8f;

#define BATCH 2
#define CH    32
#define HH    256
#define WW    512
#define LH    64
#define LW    128
#define NDIR  9
#define GPW   4   // 16-pixel groups per wave (weight-load amortization)

// leaky_relu(x, 0.01) == max(x, 0.01*x) for all finite x  ->  v_mul + v_max
__device__ __forceinline__ float leaky(float x) { return fmaxf(x, 0.01f * x); }

__device__ __forceinline__ v8f wmma_f16(v16h a, v16h b, v8f c) {
  // D = A(16xK f16) * B(Kx16 f16) + C(16x16 f32), K=32
  return __builtin_amdgcn_wmma_f32_16x16x32_f16(false, a, false, b, (short)0, c,
                                                false, false);
}

// Build an A-matrix tile (16x32, f16) from a row-major f32 weight [nRows][stride],
// taking rows [rowOff, rowOff+16) and columns [colOff, colOff+validK), zero-padded.
// A order used here (contiguous k, same as the B tiles): rows across lanes (M = lane&15); VGPR j holds
// K = {2j,2j+1}, +16 for lanes 16-31.
__device__ __forceinline__ v16h load_a_tile(const float* w, int stride, int rowOff,
                                            int nRows, int colOff, int validK,
                                            int lane) {
  const int m  = lane & 15;
  const int hi = lane >> 4;
  const int row = rowOff + m;
  v16h a;
#pragma unroll
  for (int j = 0; j < 8; ++j) {
    const int kb = 2 * j + hi * 16;
    float f0 = (row < nRows && kb     < validK) ? w[row * stride + colOff + kb]     : 0.f;
    float f1 = (row < nRows && kb + 1 < validK) ? w[row * stride + colOff + kb + 1] : 0.f;
    a[2 * j]     = (_Float16)f0;
    a[2 * j + 1] = (_Float16)f1;
  }
  return a;
}

__global__ __launch_bounds__(256) void erc_mapping_wmma_kernel(
    const float* __restrict__ lr, const float* __restrict__ hr,
    const float* __restrict__ w0, const float* __restrict__ w1,
    const float* __restrict__ w2, const float* __restrict__ w3,
    float* __restrict__ out) {
  // Per-wave staging buffer for inter-layer relayout (f16): 768 halfs/wave.
  __shared__ _Float16 stage_all[8][768];

  const int tid  = threadIdx.x;
  const int wave = tid >> 5;
  const int lane = tid & 31;
  const int p    = lane & 15;   // pixel within the 16-px group (N index)
  const int hi   = lane >> 4;   // upper lane-half flag

  // 4096 waves; each wave handles 4 consecutive 16-px groups of one row.
  // b in [0,2), y in [0,256), 8 chunks of 64 px per row.
  const int gw     = blockIdx.x * 8 + wave;
  const int b      = gw >> 11;
  const int rem    = gw & 2047;
  const int y      = rem >> 3;
  const int x0base = (rem & 7) << 6;

  _Float16* st = stage_all[wave];

  // ---- weight A-tiles (held in VGPRs for the whole wave, reused 4x) ----
  const v16h a_lr0 = load_a_tile(w0, 66, 0, 32, 0, 32, lane);   // W0[:, lr ch 0..31], rows 0..15
  const v16h a_lr1 = load_a_tile(w0, 66, 16, 32, 0, 32, lane);  // rows 16..31
  const v16h a_hr0 = load_a_tile(w0, 66, 0, 32, 32, 32, lane);  // W0[:, hr ch]
  const v16h a_hr1 = load_a_tile(w0, 66, 16, 32, 32, 32, lane);
  const v16h a_d0  = load_a_tile(w0, 66, 0, 32, 64, 2, lane);   // W0[:, distance ch]
  const v16h a_d1  = load_a_tile(w0, 66, 16, 32, 64, 2, lane);
  const v16h a_w1  = load_a_tile(w1, 32, 0, 16, 0, 32, lane);   // 16x32
  const v16h a_w2  = load_a_tile(w2, 16, 0, 8, 0, 16, lane);    // 8x16 zero-padded

  v8f zf;
#pragma unroll
  for (int i = 0; i < 8; ++i) zf[i] = 0.f;

  // Distance-pattern phase values (exact small ints, fine in f16).
  // x-phase: depends only on p (groups are 16-aligned); y-phase fixed per wave.
  const int ix = p & 3, iy = y & 3;
  const float d0_base = (ix < 2) ? (float)(ix - 2) : (float)(ix - 1); // [-2,-1,1,2]
  const float d0_fwd  = (float)(ix + 1);                              // [1,2,3,4]
  const float d0_rev  = (float)(4 - ix);                              // [4,3,2,1]
  const float d1_base = (iy < 2) ? (float)(iy - 2) : (float)(iy - 1);
  const float d1_fwd  = (float)(iy + 1);
  const float d1_rev  = (float)(4 - iy);

  // direction order: c, l, r, t, b, lt, rt, lb, rb
  const int dy4[NDIR]   = {0, 0, 0, -4, 4, -4, -4, 4, 4};
  const int dx4[NDIR]   = {0, -4, 4, 0, 0, -4, 4, -4, 4};
  const int matid[NDIR] = {0, 1, 2, 3, 4, 1, 2, 3, 4};

#pragma unroll 1
  for (int g = 0; g < GPW; ++g) {
    const int x0 = x0base + (g << 4);
    const int x  = x0 + p;

    // ---- HR B-matrix (shared by all 9 directions): ch k at K-position k ----
    v16h b_hr;
    {
      const int cb = hi << 4;
#pragma unroll
      for (int j = 0; j < 8; ++j) {
        const float* ptr = hr + (((size_t)(b * CH + cb + 2 * j)) * HH + y) * WW + x;
        b_hr[2 * j]     = (_Float16)ptr[0];
        b_hr[2 * j + 1] = (_Float16)ptr[(size_t)HH * WW];
      }
    }
    const v8f hracc0 = wmma_f16(a_hr0, b_hr, zf);
    const v8f hracc1 = wmma_f16(a_hr1, b_hr, zf);

    float res[NDIR];

#pragma unroll
    for (int d = 0; d < NDIR; ++d) {
      const int dy = dy4[d], dx = dx4[d], mat = matid[d];

      // ---- per-direction LR B-matrix (nearest-upsample at shifted position) ----
      int ly = (y + dy) >> 2; ly = min(max(ly, 0), LH - 1);
      int lx = (x + dx) >> 2; lx = min(max(lx, 0), LW - 1);
      v16h b_lr;
      {
        const int cb = hi << 4;
#pragma unroll
        for (int j = 0; j < 8; ++j) {
          const float* ptr = lr + (((size_t)(b * CH + cb + 2 * j)) * LH + ly) * LW + lx;
          b_lr[2 * j]     = (_Float16)ptr[0];
          b_lr[2 * j + 1] = (_Float16)ptr[(size_t)LH * LW];
        }
      }

      // ---- distance-channel B-matrix: only K=0,1 nonzero ----
      const float d0v = (mat == 1) ? d0_rev : (mat == 2) ? d0_fwd : d0_base;
      const float d1v = (mat == 3) ? d1_rev : (mat == 4) ? d1_fwd : d1_base;
      v16h b_d;
#pragma unroll
      for (int i = 0; i < 16; ++i) b_d[i] = (_Float16)0.f;
      b_d[0] = hi ? (_Float16)0.f : (_Float16)d0v;
      b_d[1] = hi ? (_Float16)0.f : (_Float16)d1v;

      // ---- layer 0: 32-out preactivation = W0lr*LR + W0hr*HR + W0d*D ----
      v8f acc0 = wmma_f16(a_lr0, b_lr, hracc0);
      acc0 = wmma_f16(a_d0, b_d, acc0);
      v8f acc1 = wmma_f16(a_lr1, b_lr, hracc1);
      acc1 = wmma_f16(a_d1, b_d, acc1);

      // ---- relayout D->B through LDS (apply LeakyReLU, f32->f16) ----
      // act0: ch = 16*tile + r + 8*hi, pixel = p; layout [pixel][ch0..31]
#pragma unroll
      for (int r = 0; r < 8; ++r) {
        st[p * 32 + (r + hi * 8)]      = (_Float16)leaky(acc0[r]);
        st[p * 32 + (16 + r + hi * 8)] = (_Float16)leaky(acc1[r]);
      }
      v16h b1;
#pragma unroll
      for (int i = 0; i < 16; ++i) b1[i] = st[p * 32 + (hi << 4) + i];

      // ---- layer 1: 16-out ----
      v8f accl1 = wmma_f16(a_w1, b1, zf);

#pragma unroll
      for (int r = 0; r < 8; ++r)
        st[512 + p * 16 + (r + hi * 8)] = (_Float16)leaky(accl1[r]);
      v16h b2;
#pragma unroll
      for (int i = 0; i < 16; ++i) {
        _Float16 v = st[512 + p * 16 + i];
        b2[i] = hi ? (_Float16)0.f : v;   // K=16..31 zero-padded
      }

      // ---- layer 2: 8-out (rows 8..15 of A are zero) ----
      v8f accl2 = wmma_f16(a_w2, b2, zf);

      // ---- layer 3: 1-out dot with w3 (lanes 0-15 hold ch 0..7 per VGPR) ----
      float v3 = 0.f;
#pragma unroll
      for (int r = 0; r < 8; ++r) v3 = fmaf(w3[r], leaky(accl2[r]), v3);

      bool valid = true;
      if (dx < 0) valid = valid && (x >= 4);
      if (dx > 0) valid = valid && (x < WW - 4);
      if (dy < 0) valid = valid && (y >= 4);
      if (dy > 0) valid = valid && (y < HH - 4);
      res[d] = valid ? v3 : -100.f;
    }

    // ---- softmax over the 9 directions + store (lanes 0-15 hold the pixels) ----
    if (hi == 0) {
      float mx = res[0];
#pragma unroll
      for (int d = 1; d < NDIR; ++d) mx = fmaxf(mx, res[d]);
      float e[NDIR];
      float s = 0.f;
#pragma unroll
      for (int d = 0; d < NDIR; ++d) { e[d] = __expf(res[d] - mx); s += e[d]; }
      const float inv = 1.f / s;
      const size_t obase = (((size_t)b * NDIR) * HH + y) * WW + x;
#pragma unroll
      for (int d = 0; d < NDIR; ++d)
        out[obase + (size_t)d * HH * WW] = e[d] * inv;
    }
  }
}

extern "C" void kernel_launch(void* const* d_in, const int* in_sizes, int n_in,
                              void* d_out, int out_size, void* d_ws, size_t ws_size,
                              hipStream_t stream) {
  (void)in_sizes; (void)n_in; (void)out_size; (void)d_ws; (void)ws_size;
  const float* lr = (const float*)d_in[0];   // [2,32,64,128]
  const float* hr = (const float*)d_in[1];   // [2,32,256,512]
  // d_in[2], d_in[3] (lr_feature_r / hr_feature_r) are unused by the reference
  const float* w0 = (const float*)d_in[4];   // [32,66]
  const float* w1 = (const float*)d_in[5];   // [16,32]
  const float* w2 = (const float*)d_in[6];   // [8,16]
  const float* w3 = (const float*)d_in[7];   // [1,8]
  float* out = (float*)d_out;                // [2,9,256,512]

  // 4096 waves, each covering 4x16 pixels of one row; 8 waves per block
  dim3 grid(512), block(256);
  erc_mapping_wmma_kernel<<<grid, block, 0, stream>>>(lr, hr, w0, w1, w2, w3, out);
}
